// GroupedProjectedAttention_20744692040352
// MI455X (gfx1250) — matrix operand order repaired and checked by reading, not compiled, not run on hardware
//
#include <hip/hip_runtime.h>
#include <hip/hip_bf16.h>
#include <math.h>

#define DEV __device__ __forceinline__

typedef __bf16 bf16_t;
typedef __attribute__((ext_vector_type(16))) __bf16 v16bf;
typedef __attribute__((ext_vector_type(8)))  __bf16 v8bf;
typedef __attribute__((ext_vector_type(8)))  float   v8f;
typedef __attribute__((ext_vector_type(4)))  unsigned u32x4;
typedef __attribute__((ext_vector_type(8)))  int      i32x8;
typedef __attribute__((ext_vector_type(4)))  int      i32x4;

constexpr int Bc = 8, Tc = 4096, DIMc = 1024, Gc = 8, GDc = 128, ATTc = 128, KLc = 256;
constexpr float SCALEc = 0.08838834764831845f;   // 1/sqrt(128)

constexpr size_t QKV_ELEMS = (size_t)Bc * Gc * Tc * ATTc;
constexpr size_t RED_ELEMS = (size_t)Bc * Gc * KLc * ATTc;

#if defined(__has_builtin)
#if __has_builtin(__builtin_amdgcn_tensor_load_to_lds) && \
    __has_builtin(__builtin_amdgcn_s_wait_tensorcnt)
#define USE_TDM 1
#endif
#endif
#ifndef USE_TDM
#define USE_TDM 0
#endif

// ---------- helpers ----------
DEV bf16_t f2bf(float f) {
  unsigned u = __builtin_bit_cast(unsigned, f);
  unsigned r = (u + 0x7fffu + ((u >> 16) & 1u)) >> 16;
  unsigned short h = (unsigned short)r;
  return __builtin_bit_cast(bf16_t, h);
}

DEV v8f vzero() {
  v8f r;
#pragma unroll
  for (int i = 0; i < 8; ++i) r[i] = 0.0f;
  return r;
}

// A operand (16x32 bf16), row-major LDS tile.
// lanes 0-15 -> M=lane, K {0..7,16..23}; lanes 16-31 -> M=lane-16, K {8..15,24..31}
DEV v16bf frag_A(const bf16_t* base, int stride) {
  int lane = threadIdx.x & 31;
  const bf16_t* p = base + (lane & 15) * stride + ((lane >> 4) << 3);
  v8bf lo = *(const v8bf*)p;
  v8bf hi = *(const v8bf*)(p + 16);
  v16bf r;
#pragma unroll
  for (int i = 0; i < 8; ++i) { r[i] = lo[i]; r[i + 8] = hi[i]; }
  return r;
}

// B operand (32x16 bf16), column-major (N-major) LDS tile.
DEV v16bf frag_B(const bf16_t* base, int stride) {
  int lane = threadIdx.x & 31;
  const bf16_t* p = base + (lane & 15) * stride + ((lane >> 4) << 3);
  v8bf lo = *(const v8bf*)p;
  v8bf hi = *(const v8bf*)(p + 16);
  v16bf r;
#pragma unroll
  for (int i = 0; i < 8; ++i) { r[i] = lo[i]; r[i + 8] = hi[i]; }
  return r;
}

DEV v8f wmma_bf16(v16bf a, v16bf b, v8f c) {
  return __builtin_amdgcn_wmma_f32_16x16x32_bf16(false, a, false, b, (short)0, c,
                                                 false, false);
}

#if USE_TDM
// TDM tile load: `rows` x 128 bf16, row-major global source (stride 128 elems)
// -> LDS at byte offset lds_off with per-row pad to 136 elems.
DEV void tdm_load_tile_bf16(const bf16_t* gsrc, unsigned lds_off, int rows) {
  unsigned long long ga = (unsigned long long)(size_t)gsrc;
  u32x4 g0 = {};
  g0[0] = 1u;                                  // count=1 (valid user descriptor)
  g0[1] = lds_off;                             // lds_addr (bytes)
  g0[2] = (unsigned)(ga & 0xffffffffu);        // global_addr[31:0]
  g0[3] = (unsigned)((ga >> 32) & 0x1ffffffu)  // global_addr[56:32]
        | (2u << 30);                          // type = 2 ("image")
  i32x8 g1 = {};
  g1[0] = (1 << 16)        // data_size = 2 bytes
        | (1 << 20)        // pad_enable
        | (5 << 22)        // pad_interval = 64 DWORDs (one 128-bf16 row)
        | (3 << 25);       // pad_amount   = 4 DWORDs  (-> 136-elem stride)
  g1[1] = (128 << 16);     // tensor_dim0 = 128
  g1[2] = (rows << 16);    // tensor_dim1 = rows
  g1[3] = (128 << 16);     // tile_dim0 = 128
  g1[4] = rows;            // tile_dim1 = rows
  g1[5] = 128;             // tensor_dim0_stride = 128 elements
  g1[6] = 0;
  g1[7] = 0;
  i32x4 gz = {};
#if __clang_major__ >= 23
  i32x8 gz8 = {};
  __builtin_amdgcn_tensor_load_to_lds(g0, g1, gz, gz, gz8, 0);
#else
  __builtin_amdgcn_tensor_load_to_lds(g0, g1, gz, gz, 0);
#endif
}
#endif

// =====================================================================
// Kernel 1: q/k/v = xg @ W{q,k,v}, per (b,g,64-row T tile)
// =====================================================================
// One projection; A fragments hoisted and shared across q/k/v calls.
// Wave computes 4 output tiles: mt = w>>1, nt = (w&1)*4 + i.
DEV void proj_tile(const v16bf* afs, const bf16_t* wl, bf16_t* __restrict__ dst,
                   size_t trow) {
  const int tid = threadIdx.x;
  const int w = tid >> 5, lane = tid & 31;
  const int mt = w >> 1;
  const int nt0 = (w & 1) * 4;
  v8f acc[4] = {vzero(), vzero(), vzero(), vzero()};
#pragma unroll
  for (int kk = 0; kk < 4; ++kk) {
    v16bf bfr[4];
#pragma unroll
    for (int i = 0; i < 4; ++i)
      bfr[i] = frag_B(wl + (nt0 + i) * 16 * 136 + kk * 32, 136);
#pragma unroll
    for (int i = 0; i < 4; ++i) acc[i] = wmma_bf16(afs[kk], bfr[i], acc[i]);
  }
  const int mhi = (lane >> 4) * 8, nin = lane & 15;
#pragma unroll
  for (int i = 0; i < 4; ++i)
#pragma unroll
    for (int r = 0; r < 8; ++r)
      dst[(trow + mt * 16 + mhi + r) * ATTc + (nt0 + i) * 16 + nin] =
          f2bf(acc[i][r]);
}

__global__ void gpa_qkv_kernel(const float* __restrict__ x,
                               const float* __restrict__ Wq,
                               const float* __restrict__ Wk,
                               const float* __restrict__ Wv,
                               bf16_t* __restrict__ qws,
                               bf16_t* __restrict__ kws,
                               bf16_t* __restrict__ vws) {
  extern __shared__ char smem[];
  bf16_t* xs   = (bf16_t*)smem;                // [64][136] row-major (A)
  bf16_t* wq_l = xs + 64 * 136;                // [128][136] col-major (B)
  bf16_t* wk_l = wq_l + 128 * 136;
  bf16_t* wv_l = wk_l + 128 * 136;

  const int id = blockIdx.x;
  const int tile = id & 63;            // T/64
  const int g = (id >> 6) & 7;
  const int b = id >> 9;
  const int t0 = tile * 64;
  const int tid = threadIdx.x;

  const float* Wqg = Wq + (size_t)g * GDc * ATTc;
  const float* Wkg = Wk + (size_t)g * GDc * ATTc;
  const float* Wvg = Wv + (size_t)g * GDc * GDc;

  for (int i = tid; i < GDc * ATTc; i += 256) {
    int d = i >> 7, a = i & 127;
    wq_l[a * 136 + d] = f2bf(Wqg[d * ATTc + a]);
    wk_l[a * 136 + d] = f2bf(Wkg[d * ATTc + a]);
    wv_l[a * 136 + d] = f2bf(Wvg[d * GDc + a]);
  }
  for (int i = tid; i < 64 * GDc; i += 256) {
    int t = i >> 7, d = i & 127;
    xs[t * 136 + d] = f2bf(x[((size_t)b * Tc + t0 + t) * DIMc + g * GDc + d]);
  }
  __syncthreads();

  const int w = tid >> 5;
  const int mt = w >> 1;
  v16bf afs[4];
#pragma unroll
  for (int kk = 0; kk < 4; ++kk)
    afs[kk] = frag_A(xs + mt * 16 * 136 + kk * 32, 136);

  const size_t trow = ((size_t)b * Gc + g) * Tc + t0;
  proj_tile(afs, wq_l, qws, trow);
  proj_tile(afs, wk_l, kws, trow);
  proj_tile(afs, wv_l, vws, trow);
}

// =====================================================================
// Kernel 2: k_red = Pk^T @ k, v_red = Pv^T @ v   per (b,g,which,64-KL slice)
// =====================================================================
__global__ void gpa_reduce_kernel(const float* __restrict__ Pk,
                                  const float* __restrict__ Pv,
                                  const bf16_t* __restrict__ kws,
                                  const bf16_t* __restrict__ vws,
                                  bf16_t* __restrict__ kred,
                                  bf16_t* __restrict__ vred) {
  extern __shared__ char smem[];
  bf16_t* Al = (bf16_t*)smem;    // [64][40]  A = P^T slice (kl-major, K=t)
  bf16_t* Bl = Al + 64 * 40;     // [128][40] B col-major (a-major, K=t)

  const int id = blockIdx.x;
  const int klt = id & 3;
  const int which = (id >> 2) & 1;
  const int g = (id >> 3) & 7;
  const int b = id >> 6;
  const int kl0 = klt * 64;

  const float* P = (which ? Pv : Pk) + (size_t)g * Tc * KLc;
  const bf16_t* src = (which ? vws : kws) + ((size_t)b * Gc + g) * Tc * ATTc;
  bf16_t* dst = (which ? vred : kred) + ((size_t)b * Gc + g) * KLc * ATTc;

  const int tid = threadIdx.x, w = tid >> 5;
  const int rowt = w >> 1;            // 0..3 (16 kl rows each)
  const int ct0 = (w & 1) * 4;        // 4 ATT col-tiles each

  v8f acc[4] = {vzero(), vzero(), vzero(), vzero()};

  for (int t0 = 0; t0 < Tc; t0 += 32) {
    if (t0 + 32 < Tc) {
      __builtin_prefetch(&P[(size_t)(t0 + 32) * KLc + kl0], 0, 1);
      __builtin_prefetch(&src[(size_t)(t0 + 32) * ATTc], 0, 1);
    }
    for (int i = tid; i < 32 * 64; i += 256) {
      int t = i >> 6, kl = i & 63;
      Al[kl * 40 + t] = f2bf(P[(size_t)(t0 + t) * KLc + kl0 + kl]);
    }
    for (int i = tid; i < 32 * 128; i += 256) {
      int t = i >> 7, a = i & 127;
      Bl[a * 40 + t] = src[(size_t)(t0 + t) * ATTc + a];
    }
    __syncthreads();
    v16bf af = frag_A(Al + rowt * 16 * 40, 40);
    v16bf bfr[4];
#pragma unroll
    for (int c = 0; c < 4; ++c)
      bfr[c] = frag_B(Bl + (ct0 + c) * 16 * 40, 40);
#pragma unroll
    for (int c = 0; c < 4; ++c) acc[c] = wmma_bf16(af, bfr[c], acc[c]);
    __syncthreads();
  }

  const int lane = tid & 31;
  const int mhi = (lane >> 4) * 8, nin = lane & 15;
#pragma unroll
  for (int c = 0; c < 4; ++c)
#pragma unroll
    for (int r = 0; r < 8; ++r) {
      int m = kl0 + rowt * 16 + mhi + r;
      int n = (ct0 + c) * 16 + nin;
      dst[(size_t)m * ATTc + n] = f2bf(acc[c][r]);
    }
}

// =====================================================================
// Kernel 3: fused S = q@k_red^T -> softmax -> P@v_red -> @Wo -> out
// =====================================================================
__global__ void gpa_attn_kernel(const bf16_t* __restrict__ qws,
                                const bf16_t* __restrict__ kred,
                                const bf16_t* __restrict__ vred,
                                const float* __restrict__ Wo,
                                float* __restrict__ out) {
  extern __shared__ char smem[];
  bf16_t* q_l  = (bf16_t*)smem;            // [32][136]  A
  bf16_t* Wo_l = q_l  + 32 * 136;          // [128][136] B col-major
  bf16_t* kch  = Wo_l + 128 * 136;         // [64][136]  B col-major (n=kl, k=a)
  bf16_t* vch  = kch  + 64 * 136;          // [128][72]  B col-major (n=e,  k=kl)
  bf16_t* P_l  = vch  + 128 * 72;          // [32][264]  A (probabilities)
  bf16_t* O_l  = P_l  + 32 * 264;          // [32][136]  A
  float*  S_l  = (float*)(O_l + 32 * 136); // [32][260]  fp32 logits
  float*  pmax = S_l + 32 * 260;           // [32][8]
  float*  psum = pmax + 256;               // [32][8]

  const int id = blockIdx.x;
  const int tile = id & 127;               // T/32
  const int g = (id >> 7) & 7;
  const int b = id >> 10;
  const int t0 = tile * 32;
  const int tid = threadIdx.x, w = tid >> 5, lane = tid & 31;
  const size_t bg = (size_t)b * Gc + g;

  // ---- stage q tile (TDM when available) + Wo (transposed, manual) ----
#if USE_TDM
  if (w == 0) {
    tdm_load_tile_bf16(&qws[(bg * Tc + t0) * ATTc],
                       (unsigned)((char*)q_l - (char*)smem), 32);
    __builtin_amdgcn_s_wait_tensorcnt(0);
  }
#else
  for (int i = tid; i < 32 * ATTc; i += 256) {
    int t = i >> 7, a = i & 127;
    q_l[t * 136 + a] = qws[(bg * Tc + t0 + t) * ATTc + a];
  }
#endif
  const float* Wog = Wo + (size_t)g * GDc * GDc;
  for (int i = tid; i < GDc * GDc; i += 256) {
    int e = i >> 7, f = i & 127;
    Wo_l[f * 136 + e] = f2bf(Wog[e * GDc + f]);
  }
  __syncthreads();

  // ---- Phase 1: S = q @ k_red^T (chunk KL by 64) ----
  {
    const int mt = w >> 2;       // 0..1
    const int ntl = w & 3;       // 0..3 within 64-col chunk
    for (int kc = 0; kc < KLc; kc += 64) {
#if USE_TDM
      if (w == 0) {
        tdm_load_tile_bf16(&kred[(bg * KLc + kc) * ATTc],
                           (unsigned)((char*)kch - (char*)smem), 64);
        __builtin_amdgcn_s_wait_tensorcnt(0);
      }
#else
      for (int i = tid; i < 64 * ATTc; i += 256) {
        int n = i >> 7, a = i & 127;
        kch[n * 136 + a] = kred[(bg * KLc + kc + n) * ATTc + a];
      }
#endif
      __syncthreads();
      v16bf afs[4], bfs[4];
#pragma unroll
      for (int kk = 0; kk < 4; ++kk) {
        afs[kk] = frag_A(q_l + mt * 16 * 136 + kk * 32, 136);
        bfs[kk] = frag_B(kch + ntl * 16 * 136 + kk * 32, 136);
      }
      v8f acc = vzero();
#pragma unroll
      for (int kk = 0; kk < 4; ++kk) acc = wmma_bf16(afs[kk], bfs[kk], acc);
      const int mhi = (lane >> 4) * 8, nin = lane & 15;
#pragma unroll
      for (int r = 0; r < 8; ++r)
        S_l[(mt * 16 + mhi + r) * 260 + kc + ntl * 16 + nin] = acc[r];
      __syncthreads();
    }
  }

  // ---- Phase 2: row softmax of S*SCALE (32 rows x 256, 8 threads/row) ----
  {
    const int r = tid >> 3, s = tid & 7;
    float m = -3.0e38f;
    for (int i = 0; i < 32; ++i)
      m = fmaxf(m, S_l[r * 260 + s * 32 + i] * SCALEc);
    pmax[r * 8 + s] = m;
    __syncthreads();
    float rm = -3.0e38f;
#pragma unroll
    for (int i = 0; i < 8; ++i) rm = fmaxf(rm, pmax[r * 8 + i]);
    float sum = 0.0f;
    for (int i = 0; i < 32; ++i) {
      int c = s * 32 + i;
      float e = __expf(S_l[r * 260 + c] * SCALEc - rm);
      S_l[r * 260 + c] = e;
      sum += e;
    }
    psum[r * 8 + s] = sum;
    __syncthreads();
    float rs = 0.0f;
#pragma unroll
    for (int i = 0; i < 8; ++i) rs += psum[r * 8 + i];
    float inv = 1.0f / rs;
    for (int i = 0; i < 32; ++i) {
      int c = s * 32 + i;
      P_l[r * 264 + c] = f2bf(S_l[r * 260 + c] * inv);
    }
    __syncthreads();
  }

  // ---- Phase 3: O = P @ v_red (chunk K=KL by 64) ----
  {
    const int mt = w >> 2;
    const int ntb = (w & 3) * 2;
    v8f oacc[2] = {vzero(), vzero()};
    for (int kc = 0; kc < KLc; kc += 64) {
      for (int i = tid; i < 64 * GDc; i += 256) {
        int kl = i >> 7, e = i & 127;
        vch[e * 72 + kl] = vred[(bg * KLc + kc + kl) * ATTc + e];
      }
      __syncthreads();
      v16bf afs[2], bfs[2][2];
#pragma unroll
      for (int kk = 0; kk < 2; ++kk) {
        afs[kk] = frag_A(P_l + mt * 16 * 264 + kc + kk * 32, 264);
#pragma unroll
        for (int j = 0; j < 2; ++j)
          bfs[kk][j] = frag_B(vch + (ntb + j) * 16 * 72 + kk * 32, 72);
      }
#pragma unroll
      for (int kk = 0; kk < 2; ++kk)
#pragma unroll
        for (int j = 0; j < 2; ++j)
          oacc[j] = wmma_bf16(afs[kk], bfs[kk][j], oacc[j]);
      __syncthreads();
    }
    const int mhi = (lane >> 4) * 8, nin = lane & 15;
#pragma unroll
    for (int j = 0; j < 2; ++j)
#pragma unroll
      for (int r = 0; r < 8; ++r)
        O_l[(mt * 16 + mhi + r) * 136 + (ntb + j) * 16 + nin] = f2bf(oacc[j][r]);
    __syncthreads();
  }

  // ---- Phase 4: out = O @ Wo ----
  {
    const int mt = w >> 2;
    const int ntb = (w & 3) * 2;
    v8f facc[2] = {vzero(), vzero()};
#pragma unroll
    for (int kk = 0; kk < 4; ++kk) {
      v16bf af = frag_A(O_l + mt * 16 * 136 + kk * 32, 136);
      v16bf bf0 = frag_B(Wo_l + ntb * 16 * 136 + kk * 32, 136);
      v16bf bf1 = frag_B(Wo_l + (ntb + 1) * 16 * 136 + kk * 32, 136);
      facc[0] = wmma_bf16(af, bf0, facc[0]);
      facc[1] = wmma_bf16(af, bf1, facc[1]);
    }
    const int mhi = (lane >> 4) * 8, nin = lane & 15;
#pragma unroll
    for (int j = 0; j < 2; ++j)
#pragma unroll
      for (int r = 0; r < 8; ++r) {
        int m = mt * 16 + mhi + r;
        int n = (ntb + j) * 16 + nin;
        out[((size_t)b * Tc + t0 + m) * DIMc + g * GDc + n] = facc[j][r];
      }
  }
}

// =====================================================================
// Host launcher
// =====================================================================
extern "C" void kernel_launch(void* const* d_in, const int* in_sizes, int n_in,
                              void* d_out, int out_size, void* d_ws, size_t ws_size,
                              hipStream_t stream) {
  (void)in_sizes; (void)n_in; (void)out_size; (void)ws_size;
  const float* x  = (const float*)d_in[0];
  const float* Wq = (const float*)d_in[1];
  const float* Wk = (const float*)d_in[2];
  const float* Wv = (const float*)d_in[3];
  const float* Wo = (const float*)d_in[4];
  const float* Pk = (const float*)d_in[5];
  const float* Pv = (const float*)d_in[6];
  float* out = (float*)d_out;

  bf16_t* qws  = (bf16_t*)d_ws;
  bf16_t* kws  = qws + QKV_ELEMS;
  bf16_t* vws  = kws + QKV_ELEMS;
  bf16_t* kred = vws + QKV_ELEMS;
  bf16_t* vred = kred + RED_ELEMS;

  const size_t smem1 = (size_t)(64 * 136 + 3 * 128 * 136) * sizeof(bf16_t);
  const size_t smem2 = (size_t)(64 * 40 + 128 * 40) * sizeof(bf16_t);
  const size_t smem3 = (size_t)(32 * 136 + 128 * 136 + 64 * 136 + 128 * 72 +
                                32 * 264 + 32 * 136) * sizeof(bf16_t) +
                       (size_t)(32 * 260 + 512) * sizeof(float);

  gpa_qkv_kernel<<<dim3(Bc * Gc * (Tc / 64)), dim3(256), smem1, stream>>>(
      x, Wq, Wk, Wv, qws, kws, vws);
  gpa_reduce_kernel<<<dim3(Bc * Gc * 2 * 4), dim3(256), smem2, stream>>>(
      Pk, Pv, kws, vws, kred, vred);
  gpa_attn_kernel<<<dim3(Bc * Gc * (Tc / 32)), dim3(256), smem3, stream>>>(
      qws, kred, vred, Wo, out);
}
